// CWSModel_24696061952162
// MI455X (gfx1250) — compile-verified, matrix operand order checked by reading, not run on hardware
//
#include <hip/hip_runtime.h>
#include <hip/hip_bf16.h>
#include <math.h>

// ---------------- problem constants ----------------
#define Bv 32
#define Sv 512
#define Wv 8
#define HDv 200        // per-direction hidden
#define SUBE 100
#define SUBV 100000
#define UNKI 1
#define NLAB 4
#define FEATD 500      // 2*HD + SUBE

#define KPAD 224       // 200 padded to 7*32 for bf16 WMMA K
#define NCH  7         // KPAD/32 k-chunks
#define GPAD 208       // per-gate column pad (13 tiles of 16)
#define NPAD 832       // 4*GPAD
#define NT   52        // NPAD/16 n-tiles
#define MT   (Bv*Sv/16)// 1024 m-tiles in big GEMM

typedef __attribute__((ext_vector_type(16))) __bf16 v16bf;
typedef __attribute__((ext_vector_type(8)))  float  v8f;

#if __has_builtin(__builtin_amdgcn_sched_barrier)
#define SCHED_FENCE() __builtin_amdgcn_sched_barrier(0)
#else
#define SCHED_FENCE()
#endif

// K-pair swizzle inside a bf16 WMMA fragment (per ISA 16-bit A/B layout):
// VGPR p holds K pair: p<4 -> kb+2p, else 16+kb+2(p-4); kb = (lane>=16)?8:0
__device__ __forceinline__ int frag_kpair(int p, int kb) {
    return (p < 4 ? 2 * p : 16 + 2 * (p - 4)) + kb;
}
// (hi, elem) within a lane's 16 values, given K offset kk (0..31) in chunk
__device__ __forceinline__ void frag_pos(int kk, int& hi, int& elem) {
    hi = (kk >> 3) & 1;
    int local = kk - hi * 8;          // {0..7} u {16..23}
    int p, off;
    if (local < 8) { p = local >> 1;              off = local & 1; }
    else           { p = 4 + ((local - 16) >> 1); off = (local - 16) & 1; }
    elem = 2 * p + off;
}

__device__ __forceinline__ float sigf(float x) { return 1.0f / (1.0f + __expf(-x)); }

// ---------------- prep kernels (fragment-native layouts) ----------------

// X_swz: [mtile][chunk][lane][16] bf16; A-fragment layout.
__global__ void embed_swz_kernel(const int* __restrict__ chars,
                                 const int* __restrict__ bichars,
                                 const float* __restrict__ ec,
                                 const float* __restrict__ eb,
                                 __bf16* __restrict__ X) {
    int t = blockIdx.x * blockDim.x + threadIdx.x;
    const int N = MT * NCH * 32 * 16;
    if (t >= N) return;
    int elem  = t & 15;
    int lane  = (t >> 4) & 31;
    int chunk = (t >> 9) % NCH;
    int mtile = t / (16 * 32 * NCH);
    int row = mtile * 16 + (lane & 15);
    int kb  = (lane >> 4) * 8;
    int col = chunk * 32 + frag_kpair(elem >> 1, kb) + (elem & 1);
    float v = 0.0f;
    if (col < 100)      v = ec[(size_t)chars[row]   * 100 + col];
    else if (col < 200) v = eb[(size_t)bichars[row] * 100 + (col - 100)];
    X[t] = (__bf16)v;
}

// W_swz: [chunk][ntile][lane][16] bf16; B-fragment layout, gate-padded.
// Wsrc is [4*HD=800, 200] fp32 (PyTorch layout).
__global__ void prep_weight_swz_kernel(const float* __restrict__ Wsrc,
                                       __bf16* __restrict__ WT) {
    int t = blockIdx.x * blockDim.x + threadIdx.x;
    const int N = NCH * NT * 32 * 16;
    if (t >= N) return;
    int elem  = t & 15;
    int lane  = (t >> 4) & 31;
    int ntile = (t >> 9) % NT;
    int chunk = t / (16 * 32 * NT);
    int n  = ntile * 16 + (lane & 15);
    int kb = (lane >> 4) * 8;
    int K  = chunk * 32 + frag_kpair(elem >> 1, kb) + (elem & 1);
    int gate = n / GPAD, j = n % GPAD;
    float v = 0.0f;
    if (j < HDv && K < HDv) v = Wsrc[(size_t)(gate * HDv + j) * HDv + K];
    WT[t] = (__bf16)v;
}

__global__ void prep_bias_kernel(const float* __restrict__ bsrc,
                                 float* __restrict__ bpad) {
    int n = blockIdx.x * blockDim.x + threadIdx.x;
    if (n >= NPAD) return;
    int gate = n / GPAD, j = n % GPAD;
    bpad[n] = (j < HDv) ? bsrc[gate * HDv + j] : 0.0f;
}

// ---------------- big GEMM: XG = X @ W^T + bias (time-major out) --------
// Each wave: 1 m-tile x 4 n-tiles, A fragment reused, loads double-buffered
// with sched barriers pinning next-chunk loads above current-chunk WMMAs.
// Output layout: XG[t][b][NPAD]  (time-major for the recurrence).
__global__ void gemm_xg_kernel(const __bf16* __restrict__ Xs,
                               const __bf16* __restrict__ Ws,
                               const float* __restrict__ bias,
                               float* __restrict__ XG) {
    int lane = threadIdx.x & 31;
    int wave = threadIdx.x >> 5;
    int group = blockIdx.x * (blockDim.x >> 5) + wave;
    const int NG = MT * (NT / 4);                 // 1024*13
    if (group >= NG) return;                      // whole-wave uniform
    int tm = group / (NT / 4);
    int ng = group % (NT / 4);

    const v16bf* Xf = reinterpret_cast<const v16bf*>(Xs);
    const v16bf* Wf = reinterpret_cast<const v16bf*>(Ws);

    v8f acc[4];
#pragma unroll
    for (int q = 0; q < 4; ++q) acc[q] = (v8f){};

    // prologue: chunk 0 fragments
    v16bf a_cur = Xf[(size_t)(tm * NCH + 0) * 32 + lane];
    v16bf b_cur[4];
#pragma unroll
    for (int q = 0; q < 4; ++q)
        b_cur[q] = Wf[(size_t)(0 * NT + ng * 4 + q) * 32 + lane];
    SCHED_FENCE();

#pragma unroll
    for (int c = 0; c < NCH; ++c) {
        v16bf a_nxt;
        v16bf b_nxt[4];
        if (c + 1 < NCH) {
            a_nxt = Xf[(size_t)(tm * NCH + c + 1) * 32 + lane];
#pragma unroll
            for (int q = 0; q < 4; ++q)
                b_nxt[q] = Wf[(size_t)((c + 1) * NT + ng * 4 + q) * 32 + lane];
        }
        SCHED_FENCE();   // keep next-chunk loads issued before these WMMAs
#pragma unroll
        for (int q = 0; q < 4; ++q)
            acc[q] = __builtin_amdgcn_wmma_f32_16x16x32_bf16(
                false, a_cur, false, b_cur[q], (short)0, acc[q], false, false);
        SCHED_FENCE();
        if (c + 1 < NCH) {
            a_cur = a_nxt;
#pragma unroll
            for (int q = 0; q < 4; ++q) b_cur[q] = b_nxt[q];
        }
    }

    int r = lane & 15, hi = lane >> 4;
#pragma unroll
    for (int q = 0; q < 4; ++q) {
        int tn = ng * 4 + q;
#pragma unroll
        for (int v = 0; v < 8; ++v) {
            int m  = tm * 16 + v + 8 * hi;        // m = b*S + t
            int bb = m >> 9;                      // batch
            int tt = m & (Sv - 1);                // time
            int n  = tn * 16 + r;
            XG[((size_t)tt * Bv + bb) * NPAD + n] = acc[q][v] + bias[n];
        }
    }
}

// ---------------- persistent LSTM recurrence ----------------
// grid = 2 blocks (dir 0 fwd, dir 1 bwd), block = 832 threads (26 waves).
// Each wave owns (m_tile in {0,1}, n_block in 0..12) and does 4 gate tiles.
// Issue order per step (VMEM retires in order, so order matters):
//   chunk0 frags -> xg gate loads -> pipelined chunk loop -> elementwise.
__global__ void lstm_kernel(const float* __restrict__ xg_f,
                            const float* __restrict__ xg_b,
                            const __bf16* __restrict__ whhT_f,
                            const __bf16* __restrict__ whhT_b,
                            float* __restrict__ hf,
                            float* __restrict__ hb) {
    __shared__ v16bf hbufv[2 * NCH * 32];   // [tm][chunk][lane], 32B each
    __shared__ float cbuf[Bv][GPAD];
    __bf16* hbuf = reinterpret_cast<__bf16*>(hbufv);

    const int dir = blockIdx.x;
    const float*  xg   = dir ? xg_b   : xg_f;
    const __bf16* whhT = dir ? whhT_b : whhT_f;
    float*        hs   = dir ? hb     : hf;
    const v16bf*  Wf   = reinterpret_cast<const v16bf*>(whhT);

    for (int i = threadIdx.x; i < 2 * NCH * 32 * 16; i += blockDim.x)
        hbuf[i] = (__bf16)0.0f;
    for (int i = threadIdx.x; i < Bv * GPAD; i += blockDim.x)
        cbuf[i / GPAD][i % GPAD] = 0.0f;
    __syncthreads();

    int lane = threadIdx.x & 31;
    int wave = threadIdx.x >> 5;     // 0..25
    int tm   = wave & 1;
    int nb   = wave >> 1;            // 0..12
    int r  = lane & 15;
    int hi = lane >> 4;

    for (int it = 0; it < Sv; ++it) {
        int t = dir ? (Sv - 1 - it) : it;

        v8f acc[4];
#pragma unroll
        for (int g = 0; g < 4; ++g) acc[g] = (v8f){};

        // chunk 0 fragments first (shortest wait for first WMMA)
        v16bf a_cur = hbufv[(tm * NCH + 0) * 32 + lane];
        v16bf b_cur[4];
#pragma unroll
        for (int g = 0; g < 4; ++g)
            b_cur[g] = Wf[(size_t)(0 * NT + g * (GPAD / 16) + nb) * 32 + lane];
        SCHED_FENCE();

        // xg gate addends next: they retire behind chunk0 but have the whole
        // WMMA phase to complete before the elementwise phase consumes them.
        float xgv[4][8];
#pragma unroll
        for (int v = 0; v < 8; ++v) {
            int bidx = tm * 16 + v + 8 * hi;
            int n    = nb * 16 + r;
            size_t xrow = ((size_t)t * Bv + bidx) * NPAD;
#pragma unroll
            for (int g = 0; g < 4; ++g)
                xgv[g][v] = xg[xrow + g * GPAD + n];
        }
        // prefetch next step's xg block (32*832 floats = 832 threads * 128B)
        if (it + 1 < Sv) {
            int t2 = dir ? (t - 1) : (t + 1);
            __builtin_prefetch(xg + ((size_t)t2 * Bv) * NPAD + (size_t)threadIdx.x * 32, 0, 0);
        }
        SCHED_FENCE();

        // pipelined GEMM over k-chunks: h(bf16,LDS) @ WhhT (4 gate tiles)
#pragma unroll
        for (int c = 0; c < NCH; ++c) {
            v16bf a_nxt;
            v16bf b_nxt[4];
            if (c + 1 < NCH) {
                a_nxt = hbufv[(tm * NCH + c + 1) * 32 + lane];
#pragma unroll
                for (int g = 0; g < 4; ++g)
                    b_nxt[g] = Wf[(size_t)((c + 1) * NT + g * (GPAD / 16) + nb) * 32 + lane];
            }
            SCHED_FENCE();   // next-chunk loads stay above these WMMAs
#pragma unroll
            for (int g = 0; g < 4; ++g)
                acc[g] = __builtin_amdgcn_wmma_f32_16x16x32_bf16(
                    false, a_cur, false, b_cur[g], (short)0, acc[g], false, false);
            SCHED_FENCE();
            if (c + 1 < NCH) {
                a_cur = a_nxt;
#pragma unroll
                for (int g = 0; g < 4; ++g) b_cur[g] = b_nxt[g];
            }
        }
        __syncthreads();   // all reads of hbuf done before anyone rewrites it

#pragma unroll
        for (int v = 0; v < 8; ++v) {
            int bidx = tm * 16 + v + 8 * hi;      // batch row 0..31
            int n    = nb * 16 + r;               // hidden col (padded)
            if (n < HDv) {
                float gi = acc[0][v] + xgv[0][v];
                float gf = acc[1][v] + xgv[1][v];
                float gg = acc[2][v] + xgv[2][v];
                float go = acc[3][v] + xgv[3][v];
                float c  = sigf(gf) * cbuf[bidx][n] + sigf(gi) * tanhf(gg);
                float h  = sigf(go) * tanhf(c);
                cbuf[bidx][n] = c;
                hs[((size_t)bidx * Sv + t) * HDv + n] = h;
                // write h into the swizzled A-fragment slot for next step
                int ch = n >> 5, kk = n & 31, hi2, elem;
                frag_pos(kk, hi2, elem);
                hbuf[(((bidx >> 4) * NCH + ch) * 32 + (hi2 * 16 + (bidx & 15))) * 16 + elem]
                    = (__bf16)h;
            }
        }
        __syncthreads();   // writes visible before next step's GEMM
    }
}

// ---------------- span features + subword gather + FFN ----------------
__global__ void span_ffn_kernel(const float* __restrict__ hf,
                                const float* __restrict__ hb,
                                const int* __restrict__ subwords,
                                const float* __restrict__ pre_w,
                                const float* __restrict__ sub_w,
                                const float* __restrict__ ffn_w,   // [4,500]
                                const float* __restrict__ ffn_b,
                                float* __restrict__ out) {
    const int SM2 = Sv - 2;
    int pos = blockIdx.x * blockDim.x + threadIdx.x;
    const int P = Bv * SM2 * Wv;
    if (pos >= P) return;
    int i = pos % Wv;
    int k = (pos / Wv) % SM2;
    int b = pos / (Wv * SM2);

    int end = k + i;
    float valid = (end <= Sv - 3) ? 1.0f : 0.0f;
    int e1 = end + 1; if (e1 > Sv - 1) e1 = Sv - 1;
    int e2 = end + 2; if (e2 > Sv - 1) e2 = Sv - 1;

    float acc0 = ffn_b[0], acc1 = ffn_b[1], acc2 = ffn_b[2], acc3 = ffn_b[3];
    const float* hfb = hf + (size_t)b * Sv * HDv;
    const float* hbb = hb + (size_t)b * Sv * HDv;

    for (int j = 0; j < HDv; ++j) {
        float sf = (hfb[(size_t)e1 * HDv + j] - hfb[(size_t)k * HDv + j]) * valid;
        acc0 += ffn_w[0 * FEATD + j] * sf;
        acc1 += ffn_w[1 * FEATD + j] * sf;
        acc2 += ffn_w[2 * FEATD + j] * sf;
        acc3 += ffn_w[3 * FEATD + j] * sf;
    }
    for (int j = 0; j < HDv; ++j) {
        float sb = (hbb[(size_t)(k + 1) * HDv + j] - hbb[(size_t)e2 * HDv + j]) * valid;
        acc0 += ffn_w[0 * FEATD + HDv + j] * sb;
        acc1 += ffn_w[1 * FEATD + HDv + j] * sb;
        acc2 += ffn_w[2 * FEATD + HDv + j] * sb;
        acc3 += ffn_w[3 * FEATD + HDv + j] * sb;
    }
    int sw  = subwords[pos];
    int sw2 = (sw >= SUBV) ? UNKI : sw;
    const float* pr = pre_w + (size_t)sw  * SUBE;
    const float* tr = sub_w + (size_t)sw2 * SUBE;
    for (int j = 0; j < SUBE; ++j) {
        float s = pr[j] + tr[j];
        acc0 += ffn_w[0 * FEATD + 2 * HDv + j] * s;
        acc1 += ffn_w[1 * FEATD + 2 * HDv + j] * s;
        acc2 += ffn_w[2 * FEATD + 2 * HDv + j] * s;
        acc3 += ffn_w[3 * FEATD + 2 * HDv + j] * s;
    }
    out[(size_t)pos * NLAB + 0] = acc0;
    out[(size_t)pos * NLAB + 1] = acc1;
    out[(size_t)pos * NLAB + 2] = acc2;
    out[(size_t)pos * NLAB + 3] = acc3;
}

// ---------------- launch ----------------
extern "C" void kernel_launch(void* const* d_in, const int* in_sizes, int n_in,
                              void* d_out, int out_size, void* d_ws, size_t ws_size,
                              hipStream_t stream) {
    const int*   chars    = (const int*)d_in[0];
    const int*   bichars  = (const int*)d_in[1];
    const int*   subwords = (const int*)d_in[2];
    const float* ec       = (const float*)d_in[3];
    const float* eb       = (const float*)d_in[4];
    const float* esub     = (const float*)d_in[5];
    const float* epre     = (const float*)d_in[6];
    const float* Wih_f    = (const float*)d_in[7];
    const float* Whh_f    = (const float*)d_in[8];
    const float* b_f      = (const float*)d_in[9];
    const float* Wih_b    = (const float*)d_in[10];
    const float* Whh_b    = (const float*)d_in[11];
    const float* b_b      = (const float*)d_in[12];
    const float* ffn_w    = (const float*)d_in[13];
    const float* ffn_b    = (const float*)d_in[14];
    float* out = (float*)d_out;

    size_t off = 0;
    auto carve = [&](size_t bytes) {
        void* p = (char*)d_ws + off;
        off += (bytes + 255) & ~(size_t)255;
        return p;
    };
    __bf16* x_swz   = (__bf16*)carve((size_t)MT * NCH * 32 * 16 * 2);
    __bf16* wihT_f  = (__bf16*)carve((size_t)NCH * NT * 32 * 16 * 2);
    __bf16* wihT_b  = (__bf16*)carve((size_t)NCH * NT * 32 * 16 * 2);
    __bf16* whhT_f  = (__bf16*)carve((size_t)NCH * NT * 32 * 16 * 2);
    __bf16* whhT_b  = (__bf16*)carve((size_t)NCH * NT * 32 * 16 * 2);
    float*  bp_f    = (float*)carve((size_t)NPAD * 4);
    float*  bp_b    = (float*)carve((size_t)NPAD * 4);
    float*  xg_f    = (float*)carve((size_t)Bv * Sv * NPAD * 4);
    float*  xg_b    = (float*)carve((size_t)Bv * Sv * NPAD * 4);
    float*  hf      = (float*)carve((size_t)Bv * Sv * HDv * 4);
    float*  hb      = (float*)carve((size_t)Bv * Sv * HDv * 4);

    // 1. embeddings -> bf16, fragment-swizzled A layout
    {
        int n = MT * NCH * 32 * 16;
        embed_swz_kernel<<<(n + 255) / 256, 256, 0, stream>>>(
            chars, bichars, ec, eb, x_swz);
    }
    // 2. weight repack into fragment-swizzled B layout; bias pad
    {
        int n = NCH * NT * 32 * 16;
        int blk = (n + 255) / 256;
        prep_weight_swz_kernel<<<blk, 256, 0, stream>>>(Wih_f, wihT_f);
        prep_weight_swz_kernel<<<blk, 256, 0, stream>>>(Wih_b, wihT_b);
        prep_weight_swz_kernel<<<blk, 256, 0, stream>>>(Whh_f, whhT_f);
        prep_weight_swz_kernel<<<blk, 256, 0, stream>>>(Whh_b, whhT_b);
        prep_bias_kernel<<<(NPAD + 255) / 256, 256, 0, stream>>>(b_f, bp_f);
        prep_bias_kernel<<<(NPAD + 255) / 256, 256, 0, stream>>>(b_b, bp_b);
    }
    // 3. input projections (WMMA GEMMs): xg = x @ WihT + b, time-major out
    {
        int groups = MT * (NT / 4);               // 1024*13
        int blocks = groups / 8;                  // 8 waves per 256-thr block
        gemm_xg_kernel<<<blocks, 256, 0, stream>>>(x_swz, wihT_f, bp_f, xg_f);
        gemm_xg_kernel<<<blocks, 256, 0, stream>>>(x_swz, wihT_b, bp_b, xg_b);
    }
    // 4. persistent bidirectional LSTM recurrence (WMMA per step)
    lstm_kernel<<<2, 832, 0, stream>>>(xg_f, xg_b, whhT_f, whhT_b, hf, hb);

    // 5. span features + subword gather + FFN
    {
        int P = Bv * (Sv - 2) * Wv;
        span_ffn_kernel<<<(P + 255) / 256, 256, 0, stream>>>(
            hf, hb, subwords, epre, esub, ffn_w, ffn_b, out);
    }
}
